// TemporalAttention_30554397344484
// MI455X (gfx1250) — matrix operand order repaired and checked by reading, not compiled, not run on hardware
//
#include <hip/hip_runtime.h>

// ---------------------------------------------------------------------------
// TemporalAttention for MI455X (gfx1250): bf16 WMMA pipeline
//   x[16,4096,256] -> windows [256, 256, 256]
//   1) qkv  = x @ Wqkv            (WMMA bf16, out bf16 in ws)
//   2) attn = banded softmax(qk^T)v per (window, head)   (WMMA bf16)
//   3) out  = attn @ Wproj + b    (WMMA bf16, out fp32)
// bf16 tiles are staged to LDS with GLOBAL_LOAD_ASYNC_TO_LDS_B128 and
// double-buffered so the ASYNC DMA overlaps the WMMA pipe (ASYNCcnt waits).
// ---------------------------------------------------------------------------

#define WIN       256
#define HEADS     4
#define HEAD_DIM  64
#define RADIUS    32
#define C_DIM     256
#define NEG_INF   (-1e30f)
#define SCALE_F   0.125f            // 64^-0.5

typedef __attribute__((ext_vector_type(16))) __bf16 v16bf;
typedef __attribute__((ext_vector_type(8)))  __bf16 v8bf;
typedef __attribute__((ext_vector_type(8)))  float  v8f;

static __device__ __forceinline__ unsigned short bfbits(float f) {
  union { float f; unsigned u; } a; a.f = f;
  unsigned r = a.u + 0x7FFFu + ((a.u >> 16) & 1u);   // round-to-nearest-even
  return (unsigned short)(r >> 16);
}
static __device__ __forceinline__ __bf16 f2bf(float f) {
  unsigned short h = bfbits(f);
  return __builtin_bit_cast(__bf16, h);
}
static __device__ __forceinline__ unsigned pk2(float a, float b) {
  return (unsigned)bfbits(a) | ((unsigned)bfbits(b) << 16);
}

// CDNA5 async DMA: copy 16 bytes global -> LDS, tracked on ASYNCcnt.
// LDS dest address = low 32 bits of the generic pointer (LDS aperture rule).
static __device__ __forceinline__ void async_copy16(void* ldsDst, const void* gsrc) {
  unsigned lds = (unsigned)(size_t)ldsDst;
  unsigned long long ga = (unsigned long long)(size_t)gsrc;
  asm volatile("global_load_async_to_lds_b128 %0, %1, off"
               :: "v"(lds), "v"(ga) : "memory");
}
template <int N>
static __device__ __forceinline__ void async_wait() {
  asm volatile("s_wait_asynccnt %0" :: "n"(N) : "memory");
}

// Build a 16x32 bf16 A-fragment, contiguous k order:
// lane half h holds K[16h..16h+15] (k0 = base + half*8 -> start base + half*16).
static __device__ __forceinline__ v16bf make_afrag(const __bf16* rowPtr, int k0) {
  v8bf c1 = *(const v8bf*)(rowPtr + k0 + (k0 & 8));
  v8bf c2 = *(const v8bf*)(rowPtr + k0 + (k0 & 8) + 8);
  v16bf a;
#pragma unroll
  for (int i = 0; i < 8; ++i) { a[i] = c1[i]; a[8 + i] = c2[i]; }
  return a;
}

// ---------------------------------------------------------------------------
// Weight transpose + fp32->bf16:  Wt[n*K + k] = bf16(W[k*N + n])
// ---------------------------------------------------------------------------
__global__ void __launch_bounds__(256)
transpose_to_bf16(const float* __restrict__ W, __bf16* __restrict__ Wt, int K, int N) {
  int idx = blockIdx.x * 256 + threadIdx.x;
  if (idx >= K * N) return;
  int k = idx / N, n = idx - k * N;
  Wt[(size_t)n * K + k] = f2bf(W[(size_t)k * N + n]);
}

// ---------------------------------------------------------------------------
// Tile staging: fp32 source converts through VGPRs (+ prefetch of the step
// after next); bf16 source rides the async DMA engine.
// ---------------------------------------------------------------------------
__device__ __forceinline__ void stage16(__bf16* dst, const float* src) {
  const float4* s4 = (const float4*)src;
  float4 v0 = s4[0], v1 = s4[1], v2 = s4[2], v3 = s4[3];
  uint4 w0, w1;
  w0.x = pk2(v0.x, v0.y); w0.y = pk2(v0.z, v0.w);
  w0.z = pk2(v1.x, v1.y); w0.w = pk2(v1.z, v1.w);
  w1.x = pk2(v2.x, v2.y); w1.y = pk2(v2.z, v2.w);
  w1.z = pk2(v3.x, v3.y); w1.w = pk2(v3.z, v3.w);
  *(uint4*)dst = w0; *(uint4*)(dst + 8) = w1;
  __builtin_prefetch(src + 32, 0, 1);   // keep HBM stream ahead
}
__device__ __forceinline__ void stage16(__bf16* dst, const __bf16* src) {
  async_copy16(dst, src);
  async_copy16(dst + 8, src + 8);
}

// ---------------------------------------------------------------------------
// GEMM: C[M x N] = A[M x 256] @ Wt^T  (Wt stored [N x 256] bf16)
// Block: 256 threads = 8 waves, tile 128(M) x 64(N), K step 32,
// double-buffered LDS with async prefetch of the next K step.
// ---------------------------------------------------------------------------
template <typename AT, bool OUTF32>
__global__ void __launch_bounds__(256)
gemm_bf16_kernel(const AT* __restrict__ A, const __bf16* __restrict__ Bt,
                 const float* __restrict__ bias, void* __restrict__ out, int N) {
  const int tid = threadIdx.x;
  const int wave = tid >> 5, lane = tid & 31, lane15 = lane & 15, hf = lane >> 4;
  const int Mtile = blockIdx.x * 128;
  const int Ntile = blockIdx.y * 64;

  __shared__ __align__(64) __bf16 As[2][128][32];
  __shared__ __align__(64) __bf16 Bs[2][64][32];

  const int arow = tid >> 1, aseg = (tid & 1) * 16;
  const int brow = tid >> 2, bseg = (tid & 3) * 8;
  const AT*     Ap = A  + (size_t)(Mtile + arow) * 256 + aseg;
  const __bf16* Bp = Bt + (size_t)(Ntile + brow) * 256 + bseg;

  // async instructions per thread per K step (A: 2 if bf16, B: 1)
  constexpr int NPS = (sizeof(AT) == 2) ? 3 : 1;

  // preload K step 0 into buffer 0
  stage16(&As[0][arow][aseg], Ap);
  async_copy16(&Bs[0][brow][bseg], Bp);

  v8f acc[4] = {};

#pragma unroll
  for (int ks = 0; ks < 8; ++ks) {
    const int cur = ks & 1;
    if (ks < 7) {
      // issue next step's tiles into the other buffer (readers of that
      // buffer finished at the barrier closing step ks-1)
      stage16(&As[cur ^ 1][arow][aseg], Ap + (ks + 1) * 32);
      async_copy16(&Bs[cur ^ 1][brow][bseg], Bp + (ks + 1) * 32);
      async_wait<NPS>();   // older batch (step ks) complete; next stays in flight
    } else {
      async_wait<0>();
    }
    __syncthreads();

    v16bf a = make_afrag(&As[cur][wave * 16 + lane15][0], hf * 8);
#pragma unroll
    for (int nt = 0; nt < 4; ++nt) {
      v16bf b = *(const v16bf*)&Bs[cur][nt * 16 + lane15][hf * 16];
      acc[nt] = __builtin_amdgcn_wmma_f32_16x16x32_bf16(
          false, a, false, b, (short)0, acc[nt], false, false);
    }
    __syncthreads();   // protects the buffer the step-ks+2 prefetch will write
  }

#pragma unroll
  for (int nt = 0; nt < 4; ++nt) {
    int col = Ntile + nt * 16 + lane15;
#pragma unroll
    for (int r = 0; r < 8; ++r) {
      int row = Mtile + wave * 16 + hf * 8 + r;
      float v = acc[nt][r];
      if constexpr (OUTF32) {
        ((float*)out)[(size_t)row * N + col] = v + bias[col];
      } else {
        ((__bf16*)out)[(size_t)row * N + col] = f2bf(v);
      }
    }
  }
}

// ---------------------------------------------------------------------------
// Banded causal attention, one block per (window, head).
// qkv layout: [win*256 + i][ s*256 + h*64 + d ]  (s=0:Q 1:K 2:V), bf16.
// Per 16-row I-tile only J in [i0-48, i0+16) can be unmasked -> 4 J-tiles.
// ---------------------------------------------------------------------------
__global__ void __launch_bounds__(256)
attn_kernel(const __bf16* __restrict__ qkv, __bf16* __restrict__ attn_out) {
  const int wh = blockIdx.x;
  const int win = wh >> 2, h = wh & 3;
  const int tid = threadIdx.x;
  const int wave = tid >> 5, lane = tid & 31, lane15 = lane & 15, hf = lane >> 4;

  __shared__ __align__(64) __bf16 Vt[64][256];      // V transposed: [d][j]
  __shared__ __align__(64) __bf16 Pl[8][16][64];    // per-wave P staging

  const size_t base = (size_t)win * WIN * 768;
  const __bf16* Qg = qkv + base + h * 64;
  const __bf16* Kg = qkv + base + 256 + h * 64;
  const __bf16* Vg = qkv + base + 512 + h * 64;

  // stage V transposed into LDS (shared by all 8 waves)
  for (int e = tid; e < 64 * 256; e += 256) {
    int d = e >> 8, j = e & 255;
    Vt[d][j] = Vg[(size_t)j * 768 + d];
  }
  __syncthreads();

#pragma unroll
  for (int ii = 0; ii < 2; ++ii) {
    const int it = wave + ii * 8;
    const int i0 = it * 16;

    // ---- Q fragments for 2 K-steps (Dh = 64) ----
    v16bf qa[2];
#pragma unroll
    for (int ks = 0; ks < 2; ++ks)
      qa[ks] = make_afrag(Qg + (size_t)(i0 + lane15) * 768, ks * 32 + hf * 8);

    // ---- S = scale * Q K^T over 4 J-tiles, with band mask ----
    float p[4][8], rowmax[8], rowsum[8];
#pragma unroll
    for (int r = 0; r < 8; ++r) rowmax[r] = NEG_INF;

#pragma unroll
    for (int jt = 0; jt < 4; ++jt) {
      const int jbase = i0 - 48 + jt * 16;
      v8f acc = {};
      if (jbase + 15 >= 0) {
        int j = jbase + lane15;
        int jc = j < 0 ? 0 : j;
        const __bf16* krow = Kg + (size_t)jc * 768;
#pragma unroll
        for (int ks = 0; ks < 2; ++ks) {
          v16bf kf = *(const v16bf*)(krow + ks * 32 + hf * 16);
          acc = __builtin_amdgcn_wmma_f32_16x16x32_bf16(
              false, qa[ks], false, kf, (short)0, acc, false, false);
        }
      }
      const int j = jbase + lane15;
#pragma unroll
      for (int r = 0; r < 8; ++r) {
        int i = i0 + hf * 8 + r;
        bool ok = (j >= 0) && (i - j >= 0) && (i - j <= RADIUS);
        float s = ok ? acc[r] * SCALE_F : NEG_INF;
        p[jt][r] = s;
        rowmax[r] = fmaxf(rowmax[r], s);
      }
    }

    // ---- softmax over 64 local columns (16 lanes x 4 tiles) ----
#pragma unroll
    for (int r = 0; r < 8; ++r) {
      float m = rowmax[r];
      for (int mk = 1; mk < 16; mk <<= 1) m = fmaxf(m, __shfl_xor(m, mk, 16));
      rowmax[r] = m;
      rowsum[r] = 0.0f;
    }
#pragma unroll
    for (int jt = 0; jt < 4; ++jt)
#pragma unroll
      for (int r = 0; r < 8; ++r) {
        float e = __expf(p[jt][r] - rowmax[r]);
        p[jt][r] = e;
        rowsum[r] += e;
      }
#pragma unroll
    for (int r = 0; r < 8; ++r) {
      float s = rowsum[r];
      for (int mk = 1; mk < 16; mk <<= 1) s += __shfl_xor(s, mk, 16);
      rowsum[r] = 1.0f / s;
    }

    // ---- stage P (bf16) to LDS, reshape C-layout -> A-layout ----
#pragma unroll
    for (int jt = 0; jt < 4; ++jt)
#pragma unroll
      for (int r = 0; r < 8; ++r)
        Pl[wave][hf * 8 + r][jt * 16 + lane15] = f2bf(p[jt][r]);
    asm volatile("s_wait_dscnt 0x0" ::: "memory");

    // ---- O = P @ V  (K = 64 local j, N = 64 d) ----
    v8f oacc[4] = {};
#pragma unroll
    for (int ks = 0; ks < 2; ++ks) {
      v16bf pa = make_afrag(&Pl[wave][lane15][0], ks * 32 + hf * 8);
      int j0 = i0 - 48 + ks * 32 + hf * 16;
      if (j0 < 0) j0 = 0;  // masked chunks contribute zero via P
#pragma unroll
      for (int dt = 0; dt < 4; ++dt) {
        v16bf vb = *(const v16bf*)&Vt[dt * 16 + lane15][j0];
        oacc[dt] = __builtin_amdgcn_wmma_f32_16x16x32_bf16(
            false, pa, false, vb, (short)0, oacc[dt], false, false);
      }
    }

    // ---- normalize and write bf16 [65536 x 256], heads interleaved ----
#pragma unroll
    for (int dt = 0; dt < 4; ++dt)
#pragma unroll
      for (int r = 0; r < 8; ++r) {
        int i = i0 + hf * 8 + r;
        int d = dt * 16 + lane15;
        attn_out[((size_t)win * WIN + i) * 256 + h * 64 + d] =
            f2bf(oacc[dt][r] * rowsum[r]);
      }
  }
}

// ---------------------------------------------------------------------------
extern "C" void kernel_launch(void* const* d_in, const int* in_sizes, int n_in,
                              void* d_out, int out_size, void* d_ws, size_t ws_size,
                              hipStream_t stream) {
  (void)in_sizes; (void)n_in; (void)out_size; (void)ws_size;
  const float* x     = (const float*)d_in[0];   // [16,4096,256]
  const float* Wqkv  = (const float*)d_in[1];   // [256,768]
  const float* Wproj = (const float*)d_in[2];   // [256,256]
  const float* bproj = (const float*)d_in[3];   // [256]
  float* out = (float*)d_out;

  const size_t M = (size_t)16 * 4096;           // 65536 rows
  char* ws = (char*)d_ws;
  __bf16* WtQKV = (__bf16*)ws;                                  // 768*256 bf16
  __bf16* WtP   = (__bf16*)(ws + (size_t)768 * 256 * 2);        // 256*256 bf16
  __bf16* qkv   = (__bf16*)(ws + (size_t)768 * 256 * 2
                               + (size_t)256 * 256 * 2);        // M*768 bf16
  __bf16* aout  = (__bf16*)(ws + (size_t)768 * 256 * 2
                               + (size_t)256 * 256 * 2
                               + M * 768 * 2);                  // M*256 bf16

  // 1) weights: fp32 -> bf16, transposed to [N][K]
  transpose_to_bf16<<<(256 * 768) / 256, 256, 0, stream>>>(Wqkv, WtQKV, 256, 768);
  transpose_to_bf16<<<(256 * 256) / 256, 256, 0, stream>>>(Wproj, WtP, 256, 256);

  // 2) qkv = x @ Wqkv           [65536 x 768] bf16
  gemm_bf16_kernel<float, false><<<dim3(M / 128, 768 / 64), 256, 0, stream>>>(
      x, WtQKV, nullptr, qkv, 768);

  // 3) banded attention, one block per (window, head)
  attn_kernel<<<256 * HEADS, 256, 0, stream>>>(qkv, aout);

  // 4) out = attn @ Wproj + b   [65536 x 256] fp32
  gemm_bf16_kernel<__bf16, true><<<dim3(M / 128, 256 / 64), 256, 0, stream>>>(
      aout, WtP, bproj, out, 256);
}
